// MLKA_Ablation_54065048322784
// MI455X (gfx1250) — matrix operand order repaired and checked by reading, not compiled, not run on hardware
//
#include <hip/hip_runtime.h>
#include <hip/hip_bf16.h>
#include <stdint.h>

typedef __attribute__((ext_vector_type(16))) __bf16 v16bf;
typedef __attribute__((ext_vector_type(8)))  float  v8f;
typedef __attribute__((ext_vector_type(4)))  int    v4i;

#define LOG_HW 16
#define HW     65536   // 256*256

// ---------------------------------------------------------------------------
// Async global->LDS copy (GLOBAL_LOAD_ASYNC_TO_LDS_B128), signature confirmed
// by hipcc diagnostics: (v4i AS1*, v4i AS3*, imm offset, imm cpol).
// ---------------------------------------------------------------------------
#if __has_builtin(__builtin_amdgcn_global_load_async_to_lds_b128) && \
    __has_builtin(__builtin_amdgcn_s_wait_asynccnt)
#define HAVE_ASYNC_LDS 1
typedef __attribute__((address_space(1))) v4i* g_v4i_ptr;
typedef __attribute__((address_space(3))) v4i* l_v4i_ptr;
__device__ __forceinline__ void async_copy16(const void* g, void* l) {
  __builtin_amdgcn_global_load_async_to_lds_b128(
      (g_v4i_ptr)(uintptr_t)g,
      (l_v4i_ptr)(uint32_t)(uintptr_t)l,   // low 32 bits of generic LDS addr
      0, 0);
}
__device__ __forceinline__ void async_wait_all() {
  __builtin_amdgcn_s_wait_asynccnt(0);
}
#else
#define HAVE_ASYNC_LDS 0
#endif

// ---------------------------------------------------------------------------
// Tensor Data Mover: TENSOR_LOAD_TO_LDS with a 2D tile D# (ISA ch.8).
// This toolchain's 6-arg builtin (confirmed by hipcc diagnostic):
//   (uint32x4 g0, int32x8 g1, int32x4 g2, int32x4 g3, int32x8 g4, i32 cpol)
// Guarded; fallback is the async/scalar path.
// ---------------------------------------------------------------------------
#if __has_builtin(__builtin_amdgcn_tensor_load_to_lds) && \
    __has_builtin(__builtin_amdgcn_s_wait_tensorcnt)
#define HAVE_TDM 1
typedef unsigned int u32x4 __attribute__((ext_vector_type(4)));
typedef int          i32x8 __attribute__((ext_vector_type(8)));
typedef int          i32x4 __attribute__((ext_vector_type(4)));
// Load a (Wa x S) bf16 tile from a 256x256 plane at gsrc into LDS (rows
// packed contiguously, row stride Wa elements).
__device__ __forceinline__ void tdm_load_tile(const void* gsrc, void* lds,
                                              int Wa, int S) {
  const uint64_t ga = (uint64_t)(uintptr_t)gsrc;          // byte address
  u32x4 g0;
  g0[0] = 1u;                                             // count=1, user D#
  g0[1] = (uint32_t)(uintptr_t)lds;                       // lds_addr (bytes)
  g0[2] = (uint32_t)ga;                                   // global_addr[31:0]
  g0[3] = ((uint32_t)(ga >> 32) & 0x01FFFFFFu) | (2u << 30); // addr[56:32]|type=2
  i32x8 g1;
  g1[0] = 0x00010000;            // wg_mask=0, data_size=1 (2 bytes), no flags
  g1[1] = (256 & 0xFFFF) << 16;  // tensor_dim0[15:0]=256 (atomic_addr=0)
  g1[2] = (256 & 0xFFFF) << 16;  // tensor_dim0[31:16]=0, tensor_dim1[15:0]=256
  g1[3] = (Wa & 0xFFFF) << 16;   // tensor_dim1[31:16]=0, tile_dim0=Wa
  g1[4] = S & 0xFFFF;            // tile_dim1=S, tile_dim2=0
  g1[5] = 256;                   // tensor_dim0_stride = 256 elements
  g1[6] = 0;                     // tensor_dim1_stride (unused for 2D)
  g1[7] = 0;
  i32x4 gz4 = {0, 0, 0, 0};      // groups 2/3 unused for 2D tiles
  i32x8 gz8 = {0, 0, 0, 0, 0, 0, 0, 0};
  __builtin_amdgcn_tensor_load_to_lds(g0, g1, gz4, gz4, gz8, 0);
}
__device__ __forceinline__ void tdm_wait_all() {
  __builtin_amdgcn_s_wait_tensorcnt(0);
}
#else
#define HAVE_TDM 0
#endif

// ---------------------------------------------------------------------------
// K1: BatchNorm + pf 1x1 conv (64 -> 128) as WMMA GEMM. Writes bufA (first 64
// out-channels) and bufHX (last 64) as bf16.
// Block = 256 threads = 8 waves; each wave owns 16 consecutive pixels.
// ---------------------------------------------------------------------------
__global__ __launch_bounds__(256) void bn_pf_kernel(
    const float* __restrict__ x,
    const float* __restrict__ gamma, const float* __restrict__ beta,
    const float* __restrict__ mean,  const float* __restrict__ var,
    const float* __restrict__ pfw,   const float* __restrict__ pfb,
    __bf16* __restrict__ bufA, __bf16* __restrict__ bufHX)
{
  __shared__ float bnA[64], bnB[64], biasL[128];
  __shared__ __bf16 wl[128 * 64];

  const int tid = threadIdx.x;
  if (tid < 64) {
    float inv = rsqrtf(var[tid] + 1e-5f);
    float a = gamma[tid] * inv;
    bnA[tid] = a;
    bnB[tid] = beta[tid] - mean[tid] * a;
  }
  if (tid < 128) biasL[tid] = pfb[tid];
  for (int i = tid; i < 128 * 64; i += 256) wl[i] = (__bf16)pfw[i];
  __syncthreads();

  const int lane = tid & 31;
  const int wv   = tid >> 5;
  const int p0   = blockIdx.x * 128 + wv * 16;
  const int n    = p0 >> LOG_HW;
  const int off  = p0 & (HW - 1);
  const int pix  = off + (lane & 15);
  const int hi   = lane >> 4;              // 0: lanes 0-15, 1: lanes 16-31

  // B matrices (K x N = 32 x 16), two K-slices over 64 input channels.
  // Lanes 0-15 hold K=0..15 of column N=lane; lanes 16-31 hold K=16..31.
  v16bf B0, B1;
#pragma unroll
  for (int j = 0; j < 16; ++j) {
    int ch0 = hi * 16 + j;
    int ch1 = 32 + hi * 16 + j;
    float v0 = x[((n * 64 + ch0) << LOG_HW) + pix];
    float v1 = x[((n * 64 + ch1) << LOG_HW) + pix];
    B0[j] = (__bf16)(v0 * bnA[ch0] + bnB[ch0]);
    B1[j] = (__bf16)(v1 * bnA[ch1] + bnB[ch1]);
  }

#pragma unroll
  for (int t = 0; t < 8; ++t) {            // 8 tiles of 16 output channels
    const int m = t * 16 + (lane & 15);    // A row (per §7.12.2 A layout)
    v16bf A0, A1;
#pragma unroll
    for (int e = 0; e < 16; ++e) {
      int k = hi * 16 + e;                   // lanes<16: {0..15}; else +16
      A0[e] = wl[m * 64 + k];
      A1[e] = wl[m * 64 + 32 + k];
    }
    v8f acc = {};
    acc = __builtin_amdgcn_wmma_f32_16x16x32_bf16(false, A0, false, B0,
                                                  (short)0, acc, false, false);
    acc = __builtin_amdgcn_wmma_f32_16x16x32_bf16(false, A1, false, B1,
                                                  (short)0, acc, false, false);
#pragma unroll
    for (int r = 0; r < 8; ++r) {          // C/D: vgpr r -> M = r (+8 for hi)
      int oc = t * 16 + r + hi * 8;
      float v = acc[r] + biasL[oc];
      if (oc < 64)
        bufA[((n * 64 + oc) << LOG_HW) + pix] = (__bf16)v;
      else
        bufHX[((n * 64 + (oc - 64)) << LOG_HW) + pix] = (__bf16)v;
    }
  }
}

// ---------------------------------------------------------------------------
// K2: generic depthwise conv (k x k, dilation d, pad = (k/2)*d), bf16 in/out.
// 32x32 output tile per block, haloed LDS tile. Data movement paths:
//   - interior tile, dil>1  : one TDM tensor_load_to_lds (TENSORcnt)
//   - interior tile, dil==1 : async global->LDS b128 copies (ASYNCcnt)
//   - boundary tile         : guarded scalar loads with zero padding
// grid = (64 tiles, N*32 channel-images); in/out pre-offset by channel base.
// ---------------------------------------------------------------------------
__global__ __launch_bounds__(256) void dw_conv_kernel(
    const __bf16* __restrict__ in, __bf16* __restrict__ out,
    const float* __restrict__ w, const float* __restrict__ b,
    int k, int dil)
{
  const int pad = (k >> 1) * dil;
  const int S   = 32 + 2 * pad;            // tile span incl. halo (<= 64)
  const int tid = threadIdx.x;
  const int tx  = (blockIdx.x & 7) * 32;
  const int ty  = (blockIdx.x >> 3) * 32;
  const int n   = blockIdx.y >> 5;
  const int ch  = blockIdx.y & 31;

  __shared__ __bf16 tile[72 * 64];         // worst case 64 rows
  __shared__ float  wl[81];
  __shared__ float  bl;

  if (tid < k * k) wl[tid] = w[ch * k * k + tid];
  if (tid == 0) bl = b[ch];

  const int gx0   = tx - pad;
  const int gy0   = ty - pad;
  const int gx0a  = gx0 & ~7;              // align start to 8 elems (16 B)
  const int shift = gx0 - gx0a;
  const int Wa    = (S + shift + 7) & ~7;  // aligned load width (<= 64)
  const __bf16* inc = in + ((n * 64 + ch) << LOG_HW);
  const bool interior =
      (gx0a >= 0) && (gx0a + Wa <= 256) && (gy0 >= 0) && (gy0 + S <= 256);

#if HAVE_TDM
  const bool useTDM = interior && (dil > 1);
#else
  const bool useTDM = false;
#endif
  const int RS = useTDM ? Wa : 72;         // LDS row stride (elements)

#if HAVE_TDM
  if (useTDM) {
    if (tid < 32) {                        // one wave issues the DMA
      tdm_load_tile(inc + ((gy0) << 8) + gx0a, &tile[0], Wa, S);
      tdm_wait_all();
    }
  } else
#endif
#if HAVE_ASYNC_LDS
  if (interior) {
    const int cpr   = Wa >> 3;             // 16-byte chunks per row
    const int total = cpr * S;
    for (int i = tid; i < total; i += 256) {
      int row = i / cpr;
      int col = (i - row * cpr) << 3;
      async_copy16(inc + ((gy0 + row) << 8) + gx0a + col,
                   &tile[row * 72 + col]);
    }
    async_wait_all();
  } else
#endif
  {
    const int total = Wa * S;
    for (int i = tid; i < total; i += 256) {
      int row = i / Wa;
      int col = i - row * Wa;
      int gy = gy0 + row, gx = gx0a + col;
      __bf16 v = (__bf16)0.0f;
      if (gx >= 0 && gx < 256 && gy >= 0 && gy < 256) v = inc[(gy << 8) + gx];
      tile[row * 72 + col] = v;
    }
  }
  __syncthreads();

  // Speculative prefetch of the next tile row (emits global_prefetch_b8).
  __builtin_prefetch(inc + ((ty << 8) + tx + 32 * 256), 0, 1);

  __bf16* outc = out + ((n * 64 + ch) << LOG_HW);
#pragma unroll
  for (int j = 0; j < 4; ++j) {            // 1024 px / 256 threads
    int id = tid + j * 256;
    int ox = id & 31, oy = id >> 5;
    float acc = bl;
    for (int ky = 0; ky < k; ++ky) {
      int row = oy + ky * dil;
      for (int kx = 0; kx < k; ++kx) {
        int col = ox + kx * dil + shift;
        acc += wl[ky * k + kx] * (float)tile[row * RS + col];
      }
    }
    outc[((ty + oy) << 8) + tx + ox] = (__bf16)acc;
  }
}

// ---------------------------------------------------------------------------
// K3: the two pointwise 32->32 convs (WMMA, K=32 single slice) + gate multiply
// g = (pw(u) + bias) * m ; branches: 0 = lka7 (ch 0..31), 1 = lka5 (ch 32..63)
// ---------------------------------------------------------------------------
__global__ __launch_bounds__(256) void pw_gate_kernel(
    const __bf16* __restrict__ bufU, const __bf16* __restrict__ bufM,
    __bf16* __restrict__ bufG,
    const float* __restrict__ w7, const float* __restrict__ b7,
    const float* __restrict__ w5, const float* __restrict__ b5)
{
  __shared__ __bf16 wl[2][32 * 32];
  __shared__ float  bl[2][32];
  const int tid = threadIdx.x;
  for (int i = tid; i < 1024; i += 256) {
    wl[0][i] = (__bf16)w7[i];
    wl[1][i] = (__bf16)w5[i];
  }
  if (tid < 32) { bl[0][tid] = b7[tid]; bl[1][tid] = b5[tid]; }
  __syncthreads();

  const int lane = tid & 31;
  const int wv   = tid >> 5;
  const int p0   = blockIdx.x * 128 + wv * 16;
  const int n    = p0 >> LOG_HW;
  const int off  = p0 & (HW - 1);
  const int pix  = off + (lane & 15);
  const int hi   = lane >> 4;

#pragma unroll
  for (int br = 0; br < 2; ++br) {
    v16bf B;
#pragma unroll
    for (int j = 0; j < 16; ++j) {
      int ch = hi * 16 + j;
      B[j] = bufU[((n * 64 + br * 32 + ch) << LOG_HW) + pix];
    }
#pragma unroll
    for (int t = 0; t < 2; ++t) {
      v16bf A;
#pragma unroll
      for (int e = 0; e < 16; ++e) {
        int k = hi * 16 + e;
        A[e] = wl[br][(t * 16 + (lane & 15)) * 32 + k];
      }
      v8f acc = {};
      acc = __builtin_amdgcn_wmma_f32_16x16x32_bf16(false, A, false, B,
                                                    (short)0, acc, false, false);
#pragma unroll
      for (int r = 0; r < 8; ++r) {
        int oc = t * 16 + r + hi * 8;
        int idx = ((n * 64 + br * 32 + oc) << LOG_HW) + pix;
        float v = (acc[r] + bl[br][oc]) * (float)bufM[idx];
        bufG[idx] = (__bf16)v;
      }
    }
  }
}

// ---------------------------------------------------------------------------
// K4: out = pl(hx * g) * scale + shortcut, 64->64 WMMA GEMM, fp32 output.
// ---------------------------------------------------------------------------
__global__ __launch_bounds__(256) void pl_out_kernel(
    const __bf16* __restrict__ bufHX, const __bf16* __restrict__ bufG,
    const float* __restrict__ plw, const float* __restrict__ plb,
    const float* __restrict__ scale, const float* __restrict__ x,
    float* __restrict__ out)
{
  __shared__ __bf16 wl[64 * 64];
  __shared__ float  bl[64], sl[64];
  const int tid = threadIdx.x;
  for (int i = tid; i < 64 * 64; i += 256) wl[i] = (__bf16)plw[i];
  if (tid < 64) { bl[tid] = plb[tid]; sl[tid] = scale[tid]; }
  __syncthreads();

  const int lane = tid & 31;
  const int wv   = tid >> 5;
  const int p0   = blockIdx.x * 128 + wv * 16;
  const int n    = p0 >> LOG_HW;
  const int off  = p0 & (HW - 1);
  const int pix  = off + (lane & 15);
  const int hi   = lane >> 4;

  v16bf B0, B1;
#pragma unroll
  for (int j = 0; j < 16; ++j) {
    int ch0 = hi * 16 + j;
    int ch1 = 32 + hi * 16 + j;
    int i0 = ((n * 64 + ch0) << LOG_HW) + pix;
    int i1 = ((n * 64 + ch1) << LOG_HW) + pix;
    B0[j] = (__bf16)((float)bufHX[i0] * (float)bufG[i0]);
    B1[j] = (__bf16)((float)bufHX[i1] * (float)bufG[i1]);
  }

#pragma unroll
  for (int t = 0; t < 4; ++t) {
    const int m = t * 16 + (lane & 15);
    v16bf A0, A1;
#pragma unroll
    for (int e = 0; e < 16; ++e) {
      int k = hi * 16 + e;
      A0[e] = wl[m * 64 + k];
      A1[e] = wl[m * 64 + 32 + k];
    }
    v8f acc = {};
    acc = __builtin_amdgcn_wmma_f32_16x16x32_bf16(false, A0, false, B0,
                                                  (short)0, acc, false, false);
    acc = __builtin_amdgcn_wmma_f32_16x16x32_bf16(false, A1, false, B1,
                                                  (short)0, acc, false, false);
#pragma unroll
    for (int r = 0; r < 8; ++r) {
      int oc  = t * 16 + r + hi * 8;
      int idx = ((n * 64 + oc) << LOG_HW) + pix;
      out[idx] = (acc[r] + bl[oc]) * sl[oc] + x[idx];
    }
  }
}

// ---------------------------------------------------------------------------
// Launch: K1 -> 4x dw -> 2x dd -> K3 -> K4   (all on `stream`, no sync calls)
// Workspace (bf16, 33.5M elems each): A | HX | T(->G) | M | U
// ---------------------------------------------------------------------------
extern "C" void kernel_launch(void* const* d_in, const int* in_sizes, int n_in,
                              void* d_out, int out_size, void* d_ws, size_t ws_size,
                              hipStream_t stream)
{
  const float* x        = (const float*)d_in[0];
  const float* bn_gamma = (const float*)d_in[1];
  const float* bn_beta  = (const float*)d_in[2];
  const float* bn_mean  = (const float*)d_in[3];
  const float* bn_var   = (const float*)d_in[4];
  const float* scale    = (const float*)d_in[5];
  const float* lka7_dw_w = (const float*)d_in[6];
  const float* lka7_dw_b = (const float*)d_in[7];
  const float* lka7_dd_w = (const float*)d_in[8];
  const float* lka7_dd_b = (const float*)d_in[9];
  const float* lka7_pw_w = (const float*)d_in[10];
  const float* lka7_pw_b = (const float*)d_in[11];
  const float* lka5_dw_w = (const float*)d_in[12];
  const float* lka5_dw_b = (const float*)d_in[13];
  const float* lka5_dd_w = (const float*)d_in[14];
  const float* lka5_dd_b = (const float*)d_in[15];
  const float* lka5_pw_w = (const float*)d_in[16];
  const float* lka5_pw_b = (const float*)d_in[17];
  const float* x5_w = (const float*)d_in[18];
  const float* x5_b = (const float*)d_in[19];
  const float* x7_w = (const float*)d_in[20];
  const float* x7_b = (const float*)d_in[21];
  const float* pf_w = (const float*)d_in[22];
  const float* pf_b = (const float*)d_in[23];
  const float* pl_w = (const float*)d_in[24];
  const float* pl_b = (const float*)d_in[25];
  float* out = (float*)d_out;

  const size_t CH64 = (size_t)8 * 64 * HW;   // elements per 64-channel buffer
  __bf16* bufA  = (__bf16*)d_ws;
  __bf16* bufHX = bufA  + CH64;
  __bf16* bufT  = bufHX + CH64;              // dw outputs; reused as G later
  __bf16* bufM  = bufT  + CH64;              // gate conv outputs (x7|x5)
  __bf16* bufU  = bufM  + CH64;              // dd outputs
  __bf16* bufG  = bufT;                      // alias: T is dead after dd stage

  const dim3 blk(256);
  const dim3 gpix(4096);                     // 524288 px / 128 px-per-block
  const dim3 gdw(64, 256);                   // 64 tiles x (8 imgs * 32 ch)
  const size_t choff = (size_t)32 * HW;      // +32 channels within 64-ch buf

  bn_pf_kernel<<<gpix, blk, 0, stream>>>(x, bn_gamma, bn_beta, bn_mean, bn_var,
                                         pf_w, pf_b, bufA, bufHX);

  dw_conv_kernel<<<gdw, blk, 0, stream>>>(bufA,         bufT,         lka7_dw_w, lka7_dw_b, 7, 1);
  dw_conv_kernel<<<gdw, blk, 0, stream>>>(bufA + choff, bufT + choff, lka5_dw_w, lka5_dw_b, 5, 1);
  dw_conv_kernel<<<gdw, blk, 0, stream>>>(bufA,         bufM,         x7_w,      x7_b,      7, 1);
  dw_conv_kernel<<<gdw, blk, 0, stream>>>(bufA + choff, bufM + choff, x5_w,      x5_b,      5, 1);

  dw_conv_kernel<<<gdw, blk, 0, stream>>>(bufT,         bufU,         lka7_dd_w, lka7_dd_b, 9, 4);
  dw_conv_kernel<<<gdw, blk, 0, stream>>>(bufT + choff, bufU + choff, lka5_dd_w, lka5_dd_b, 7, 3);

  pw_gate_kernel<<<gpix, blk, 0, stream>>>(bufU, bufM, bufG,
                                           lka7_pw_w, lka7_pw_b,
                                           lka5_pw_w, lka5_pw_b);

  pl_out_kernel<<<gpix, blk, 0, stream>>>(bufHX, bufG, pl_w, pl_b, scale, x, out);
}
